// FeedForwardSimple_67284957659593
// MI455X (gfx1250) — 8-bit matrix operand order repaired and checked by reading, compiled, not run on hardware
//
#include <hip/hip_runtime.h>
#include <hip/hip_bf16.h>


typedef __attribute__((ext_vector_type(16))) int   v16i;
typedef __attribute__((ext_vector_type(4)))  int   v4i;
typedef __attribute__((ext_vector_type(8)))  float v8f;
typedef unsigned long long u64;

#define M_TOK   8192
#define K_FEAT  2048
#define N_HID   8192

#if defined(__has_builtin)
# if __has_builtin(__builtin_amdgcn_global_load_async_to_lds_b128) && \
     __has_builtin(__builtin_amdgcn_s_wait_asynccnt)
#  define USE_ASYNC_LDS 1
# endif
#endif
#ifndef USE_ASYNC_LDS
# define USE_ASYNC_LDS 0
#endif

typedef __attribute__((address_space(1))) v4i* as1_v4i_ptr;
typedef __attribute__((address_space(3))) v4i* as3_v4i_ptr;

__device__ __forceinline__ void cp16_to_lds(void* ldst, const void* gsrc) {
#if USE_ASYNC_LDS
  __builtin_amdgcn_global_load_async_to_lds_b128(
      (as1_v4i_ptr)gsrc, (as3_v4i_ptr)ldst, 0, 0);
#else
  *(int4*)ldst = *(const int4*)gsrc;
#endif
}

__device__ __forceinline__ void wait_async_copies() {
#if USE_ASYNC_LDS
  __builtin_amdgcn_s_wait_asynccnt(0);
#endif
}

__device__ __forceinline__ unsigned char f32_to_e4m3(float x) {
  if (!(x == x)) return 0x7F;
  unsigned u    = __float_as_uint(x);
  unsigned sign = (u >> 31) << 7;
  float    ax   = fabsf(x);
  if (ax >= 448.0f) return (unsigned char)(sign | 0x7E);
  if (ax < 0.015625f) {
    int mm = (int)(ax * 512.0f + 0.5f);
    return (unsigned char)(sign | (unsigned)mm);
  }
  int e = (int)((u >> 23) & 0xFF);
  unsigned mant = (u >> 20) & 7u;
  unsigned rest = u & 0xFFFFFu;
  unsigned code = (unsigned)(((e - 127 + 7) << 3)) | mant;
  if (rest > 0x80000u || (rest == 0x80000u && (mant & 1u))) code += 1u;
  if (code > 0x7Eu) code = 0x7Eu;
  return (unsigned char)(sign | code);
}

__device__ __forceinline__ float gelu_tanh(float g) {
  return 0.5f * g * (1.0f + tanhf(0.7978845608028654f * (g + 0.044715f * g * g * g)));
}

__global__ void quant_x_kernel(const __hip_bfloat16* __restrict__ x,
                               unsigned char* __restrict__ x8, size_t total) {
  size_t i = (size_t)blockIdx.x * blockDim.x + threadIdx.x;
  if (i < total) x8[i] = f32_to_e4m3(__bfloat162float(x[i]));
}

__global__ void quant_wT_kernel(const float* __restrict__ w,
                                unsigned char* __restrict__ w8, int K, int N) {
  size_t idx = (size_t)blockIdx.x * blockDim.x + threadIdx.x;
  size_t total = (size_t)K * (size_t)N;
  if (idx >= total) return;
  size_t n = idx / (size_t)K;
  size_t k = idx - n * (size_t)K;
  w8[idx] = f32_to_e4m3(w[k * (size_t)N + n]);
}

__device__ __forceinline__ v16i load_a_frag(const unsigned char* As, int row, int hi) {
  v16i a;
  const u64* ap = (const u64*)(As + row * 128 + hi * 16);
#pragma unroll
  for (int j = 0; j < 8; ++j) ((u64*)&a)[j] = ap[(j >> 1) * 4 + (j & 1)];
  return a;
}

__device__ __forceinline__ v16i load_b_frag(const unsigned char* Bs, int col, int hi) {
  v16i b;
  const int4* bp = (const int4*)(Bs + col * 128 + hi * 16);
#pragma unroll
  for (int j = 0; j < 4; ++j) ((int4*)&b)[j] = bp[2 * j];
  return b;
}

__device__ __forceinline__ void stage_tile(unsigned char* lds,
                                           const unsigned char* g,
                                           int row0, int K, int kb, int tid) {
#pragma unroll
  for (int i = 0; i < 4; ++i) {
    int off = tid * 16 + i * 4096;
    int r = off >> 7, c = off & 127;
    cp16_to_lds(lds + off, g + (size_t)(row0 + r) * K + kb + c);
  }
}

__global__ __launch_bounds__(256) void geglu_gate_up_kernel(
    const unsigned char* __restrict__ x8,
    const unsigned char* __restrict__ w0,
    const unsigned char* __restrict__ w1,
    unsigned char* __restrict__ act8,
    int M, int K, int N) {
  extern __shared__ __align__(16) unsigned char smem[];
  unsigned char* As  = smem;
  unsigned char* B0s = smem + 32768;
  unsigned char* B1s = smem + 65536;

  const int tid   = threadIdx.x;
  const int lane  = tid & 31;
  const int wave  = tid >> 5;
  const int waveM = wave >> 1;
  const int waveN = wave & 1;
  const int ln    = lane & 15;
  const int hi    = lane >> 4;
  const int m0    = blockIdx.y * 128;
  const int n0    = blockIdx.x * 128;

  const v8f z8 = {0.f, 0.f, 0.f, 0.f, 0.f, 0.f, 0.f, 0.f};
  v8f accG[2][4], accU[2][4];
#pragma unroll
  for (int mt = 0; mt < 2; ++mt)
#pragma unroll
    for (int nt = 0; nt < 4; ++nt) { accG[mt][nt] = z8; accU[mt][nt] = z8; }

  stage_tile(As,  x8, m0, K, 0, tid);
  stage_tile(B0s, w0, n0, K, 0, tid);
  stage_tile(B1s, w1, n0, K, 0, tid);
  wait_async_copies();
  __syncthreads();

  for (int kb = 0; kb < K; kb += 128) {
    const int cur = (kb >> 7) & 1;
    const int curoff = cur * 16384;
    const int nxtoff = (cur ^ 1) * 16384;

    if (kb + 128 < K) {
      stage_tile(As  + nxtoff, x8, m0, K, kb + 128, tid);
      stage_tile(B0s + nxtoff, w0, n0, K, kb + 128, tid);
      stage_tile(B1s + nxtoff, w1, n0, K, kb + 128, tid);
    }

    v16i a[2];
#pragma unroll
    for (int mt = 0; mt < 2; ++mt)
      a[mt] = load_a_frag(As + curoff, waveM * 32 + mt * 16 + ln, hi);

#pragma unroll
    for (int nt = 0; nt < 4; ++nt) {
      const int col = waveN * 64 + nt * 16 + ln;
      v16i b0 = load_b_frag(B0s + curoff, col, hi);
      v16i b1 = load_b_frag(B1s + curoff, col, hi);
#pragma unroll
      for (int mt = 0; mt < 2; ++mt) {
        accG[mt][nt] = __builtin_amdgcn_wmma_f32_16x16x128_fp8_fp8(
            a[mt], b0, (short)0, accG[mt][nt], false, false);
        accU[mt][nt] = __builtin_amdgcn_wmma_f32_16x16x128_fp8_fp8(
            a[mt], b1, (short)0, accU[mt][nt], false, false);
      }
    }

    wait_async_copies();
    __syncthreads();
  }

#pragma unroll
  for (int mt = 0; mt < 2; ++mt)
#pragma unroll
    for (int nt = 0; nt < 4; ++nt) {
      const int col = n0 + waveN * 64 + nt * 16 + ln;
#pragma unroll
      for (int r = 0; r < 8; ++r) {
        const int row = m0 + waveM * 32 + mt * 16 + r + 8 * hi;
        float v = gelu_tanh(accG[mt][nt][r]) * accU[mt][nt][r];
        act8[(size_t)row * N + col] = f32_to_e4m3(v);
      }
    }
}

__global__ __launch_bounds__(256) void geglu_out_kernel(
    const unsigned char* __restrict__ a8,
    const unsigned char* __restrict__ wl,
    __hip_bfloat16* __restrict__ out,
    int M, int K, int N) {
  extern __shared__ __align__(16) unsigned char smem[];
  unsigned char* As = smem;
  unsigned char* Bs = smem + 32768;

  const int tid   = threadIdx.x;
  const int lane  = tid & 31;
  const int wave  = tid >> 5;
  const int waveM = wave >> 1;
  const int waveN = wave & 1;
  const int ln    = lane & 15;
  const int hi    = lane >> 4;
  const int m0    = blockIdx.y * 128;
  const int n0    = blockIdx.x * 128;

  const v8f z8 = {0.f, 0.f, 0.f, 0.f, 0.f, 0.f, 0.f, 0.f};
  v8f acc[2][4];
#pragma unroll
  for (int mt = 0; mt < 2; ++mt)
#pragma unroll
    for (int nt = 0; nt < 4; ++nt) acc[mt][nt] = z8;

  stage_tile(As, a8, m0, K, 0, tid);
  stage_tile(Bs, wl, n0, K, 0, tid);
  wait_async_copies();
  __syncthreads();

  for (int kb = 0; kb < K; kb += 128) {
    const int cur = (kb >> 7) & 1;
    const int curoff = cur * 16384;
    const int nxtoff = (cur ^ 1) * 16384;

    if (kb + 128 < K) {
      stage_tile(As + nxtoff, a8, m0, K, kb + 128, tid);
      stage_tile(Bs + nxtoff, wl, n0, K, kb + 128, tid);
    }

    v16i a[2];
#pragma unroll
    for (int mt = 0; mt < 2; ++mt)
      a[mt] = load_a_frag(As + curoff, waveM * 32 + mt * 16 + ln, hi);

#pragma unroll
    for (int nt = 0; nt < 4; ++nt) {
      v16i b = load_b_frag(Bs + curoff, waveN * 64 + nt * 16 + ln, hi);
#pragma unroll
      for (int mt = 0; mt < 2; ++mt)
        acc[mt][nt] = __builtin_amdgcn_wmma_f32_16x16x128_fp8_fp8(
            a[mt], b, (short)0, acc[mt][nt], false, false);
    }

    wait_async_copies();
    __syncthreads();
  }

#pragma unroll
  for (int mt = 0; mt < 2; ++mt)
#pragma unroll
    for (int nt = 0; nt < 4; ++nt) {
      const int col = n0 + waveN * 64 + nt * 16 + ln;
#pragma unroll
      for (int r = 0; r < 8; ++r) {
        const int row = m0 + waveM * 32 + mt * 16 + r + 8 * hi;
        out[(size_t)row * N + col] = __float2bfloat16(acc[mt][nt][r]);
      }
    }
}

extern "C" void kernel_launch(void* const* d_in, const int* in_sizes, int n_in,
                              void* d_out, int out_size, void* d_ws, size_t ws_size,
                              hipStream_t stream) {
  const __hip_bfloat16* x  = (const __hip_bfloat16*)d_in[0];
  const float* w_gating    = (const float*)d_in[1];
  const float* w_linear    = (const float*)d_in[2];
  __hip_bfloat16* out      = (__hip_bfloat16*)d_out;

  unsigned char* ws  = (unsigned char*)d_ws;
  const size_t SZ16M = (size_t)16 * 1024 * 1024;
  unsigned char* x8   = ws;
  unsigned char* wg0  = ws + 1 * SZ16M;
  unsigned char* wg1  = ws + 2 * SZ16M;
  unsigned char* wl8  = ws + 3 * SZ16M;
  unsigned char* act8 = ws + 4 * SZ16M;

  {
    size_t nx = (size_t)M_TOK * K_FEAT;
    quant_x_kernel<<<(unsigned)((nx + 255) / 256), 256, 0, stream>>>(x, x8, nx);
    size_t nw = (size_t)K_FEAT * N_HID;
    unsigned gw = (unsigned)((nw + 255) / 256);
    quant_wT_kernel<<<gw, 256, 0, stream>>>(w_gating,                          wg0, K_FEAT, N_HID);
    quant_wT_kernel<<<gw, 256, 0, stream>>>(w_gating + (size_t)K_FEAT * N_HID, wg1, K_FEAT, N_HID);
    quant_wT_kernel<<<gw, 256, 0, stream>>>(w_linear,                          wl8, N_HID, K_FEAT);
  }

  {
    dim3 grid(N_HID / 128, M_TOK / 128);
    geglu_gate_up_kernel<<<grid, 256, 96 * 1024, stream>>>(x8, wg0, wg1, act8,
                                                           M_TOK, K_FEAT, N_HID);
  }

  {
    dim3 grid(K_FEAT / 128, M_TOK / 128);
    geglu_out_kernel<<<grid, 256, 64 * 1024, stream>>>(act8, wl8, out,
                                                       M_TOK, N_HID, K_FEAT);
  }
}
